// LocalWindowAttention_56100862820514
// MI455X (gfx1250) — matrix operand order repaired and checked by reading, not compiled, not run on hardware
//
#include <hip/hip_runtime.h>

// Local window attention, fully fused: one workgroup per 16-token window.
// B=2, S=8192, DIM=1024, HEADS=16, HD=64, WINDOW=16 -> 1024 blocks.
// bf16 WMMA (v_wmma_f32_16x16x32_bf16) with fp32 accumulation everywhere.
// Weights are pre-converted fp32->bf16 once into d_ws by a pre-pass kernel,
// so the fused GEMMs consume bf16 B-fragments straight from global/L2.

#define DIMC   1024
#define HEADS  16
#define HD     64
#define WIN    16

typedef __attribute__((ext_vector_type(16))) __bf16       v16bf;
typedef __attribute__((ext_vector_type(8)))  float        v8f;
typedef __attribute__((ext_vector_type(4)))  unsigned int u32x4;
typedef unsigned short u16;

union Frag {
    v16bf        bf;
    u32x4        q[2];
    unsigned int u[8];
};

// round-to-nearest-even f32 -> bf16
__device__ __forceinline__ u16 f2bf(float f) {
    unsigned int a = __float_as_uint(f);
    a = a + 0x7FFFu + ((a >> 16) & 1u);
    return (u16)(a >> 16);
}
__device__ __forceinline__ unsigned int pack_bf16(float lo, float hi) {
    unsigned int a = __float_as_uint(lo);
    unsigned int b = __float_as_uint(hi);
    a = a + 0x7FFFu + ((a >> 16) & 1u);
    b = b + 0x7FFFu + ((b >> 16) & 1u);
    return (a >> 16) | (b & 0xFFFF0000u);
}

__device__ __forceinline__ v8f wmma_bf16(const Frag& a, const Frag& b, v8f c) {
    return __builtin_amdgcn_wmma_f32_16x16x32_bf16(
        /*neg_a=*/false, a.bf, /*neg_b=*/false, b.bf,
        /*c_mod=*/(short)0, c, /*reuse_a=*/false, /*reuse_b=*/false);
}

// ---- pre-pass: fp32 weight matrix -> bf16 (packed pairs) ----
__global__ __launch_bounds__(256)
void cvt_w_kernel(const float* __restrict__ src, unsigned int* __restrict__ dst,
                  int npairs) {
    const int i = blockIdx.x * 256 + threadIdx.x;
    if (i < npairs) {
        const float* p = src + (size_t)i * 2;
        dst[i] = pack_bf16(p[0], p[1]);
    }
}

// C[16 x 128] += A[16 x 1024](bf16, LDS, row-major stride DIMC) * W^T,
// W is [1024 x 1024] bf16 row-major in global (row n = output col, K contiguous,
// L2-resident). Per wave: 8 accumulator tiles of 16x16, K-loop in steps of 32.
__device__ __forceinline__ void gemm_16x128(const u16* __restrict__ Alds,
                                            const u16* __restrict__ Wb,
                                            int nBase, int ln, int hi, v8f acc[8]) {
    for (int kc = 0; kc < DIMC; kc += 32) {
        Frag a;
        const u16* ab = Alds + ln * DIMC + kc + hi * 8;
        a.q[0] = *(const u32x4*)(ab);        // K 0..7  (+8 for hi lanes)
        a.q[1] = *(const u32x4*)(ab + 16);   // K 16..23 (+8 for hi lanes)
#pragma unroll
        for (int t = 0; t < 8; ++t) {
            const u16* wb = Wb + (size_t)(nBase + t * 16 + ln) * DIMC + kc + hi * 8;
            Frag b;
            b.q[0] = *(const u32x4*)(wb);    // K rel 0..7  (+8 for hi lanes)
            b.q[1] = *(const u32x4*)(wb + 16);// K rel 16..23 (+8 for hi lanes)
            acc[t] = wmma_bf16(a, b, acc[t]);
        }
    }
}

// Split-half RoPE applied directly on accumulator tiles (wave owns 2 heads =
// tiles 0..3 and 4..7; pair (d, d+32) is same lane, +2 tiles), then bf16->LDS.
__device__ __forceinline__ void rope_store(v8f acc[8], u16* __restrict__ dst,
                                           const float* __restrict__ Cs,
                                           const float* __restrict__ Ss,
                                           int nBase, int ln, int hi) {
#pragma unroll
    for (int e = 0; e < 2; ++e) {
        const int t0 = e * 4;
#pragma unroll
        for (int j = 0; j < 2; ++j) {
            const int jidx = j * 16 + ln;   // 0..31 (rotation half index)
#pragma unroll
            for (int v = 0; v < 8; ++v) {
                const int m = v + hi * 8;   // token position in window
                const float c = Cs[m * 32 + jidx];
                const float s = Ss[m * 32 + jidx];
                const float a = acc[t0 + j][v];
                const float b = acc[t0 + j + 2][v];
                acc[t0 + j][v]     = a * c - b * s;
                acc[t0 + j + 2][v] = a * s + b * c;
            }
        }
    }
#pragma unroll
    for (int t = 0; t < 8; ++t) {
#pragma unroll
        for (int v = 0; v < 8; ++v) {
            const int m = v + hi * 8;
            dst[m * DIMC + nBase + t * 16 + ln] = f2bf(acc[t][v]);
        }
    }
}

__global__ __launch_bounds__(256)
void lwa_fused_kernel(const float* __restrict__ x,
                      const float* __restrict__ rope,
                      const u16* __restrict__ wq,
                      const u16* __restrict__ wk,
                      const u16* __restrict__ wv,
                      const u16* __restrict__ wo,
                      float* __restrict__ out) {
    extern __shared__ unsigned char smem[];
    u16*   Xs = (u16*)smem;              // [16][1024] x window bf16 (reused as attn)
    u16*   Qs = Xs + WIN * DIMC;         // [16][1024] q post-RoPE
    u16*   Ks = Qs + WIN * DIMC;         // [16][1024] k post-RoPE
    u16*   Vt = Ks + WIN * DIMC;         // [head][d(64)][key(16)] transposed v
    u16*   Ps = Vt + WIN * DIMC;         // [8 waves][16][16] P bounce buffer
    float* Cs = (float*)(Ps + 8 * 256);  // [16][32] cos table
    float* Ss = Cs + 512;                // [16][32] sin table

    const int tid  = threadIdx.x;
    const int lane = tid & 31;
    const int wid  = tid >> 5;
    const int hi   = (lane >> 4) & 1;
    const int ln   = lane & 15;
    const int tokenBase = blockIdx.x * WIN;
    const int nBase     = wid * 128;     // wave's output-column slice (2 heads)

    // ---- phase 0: x window -> LDS (bf16) + RoPE cos/sin tables ----
    for (int i = tid; i < (WIN * DIMC) / 2; i += 256) {
        const int row = i >> 9;          // / (DIMC/2)
        const int c2  = i & 511;
        const float* px = x + (size_t)(tokenBase + row) * DIMC + c2 * 2;
        ((unsigned int*)Xs)[row * 512 + c2] = pack_bf16(px[0], px[1]);
    }
    for (int i = tid; i < WIN * 32; i += 256) {
        const float a = rope[i];         // rope_freqs[m][j], m<16, row stride 32
        Cs[i] = cosf(a);
        Ss[i] = sinf(a);
    }
    __syncthreads();

    v8f acc[8];

    // ---- phase 1: Q = x @ wq^T, RoPE, -> Qs ----
#pragma unroll
    for (int t = 0; t < 8; ++t) acc[t] = {};
    gemm_16x128(Xs, wq, nBase, ln, hi, acc);
    rope_store(acc, Qs, Cs, Ss, nBase, ln, hi);

    // ---- phase 2: K = x @ wk^T, RoPE, -> Ks ----
#pragma unroll
    for (int t = 0; t < 8; ++t) acc[t] = {};
    gemm_16x128(Xs, wk, nBase, ln, hi, acc);
    rope_store(acc, Ks, Cs, Ss, nBase, ln, hi);

    // ---- phase 3: V = x @ wv^T -> Vt (per-head transposed [d][key]) ----
#pragma unroll
    for (int t = 0; t < 8; ++t) acc[t] = {};
    gemm_16x128(Xs, wv, nBase, ln, hi, acc);
#pragma unroll
    for (int t = 0; t < 8; ++t) {
        const int h  = wid * 2 + (t >> 2);
        const int dl = (t & 3) * 16 + ln;
#pragma unroll
        for (int v = 0; v < 8; ++v) {
            const int key = v + hi * 8;
            Vt[(h * HD + dl) * WIN + key] = f2bf(acc[t][v]);
        }
    }
    __syncthreads();  // all waves done reading Xs; attn result will alias it

    // ---- phase 4: attention for this wave's 2 heads ----
    u16* As = Xs;  // reuse x-window LDS for attention output (bf16)
#pragma unroll
    for (int e = 0; e < 2; ++e) {
        const int h = wid * 2 + e;
        // scores = Q K^T : M=16 N=16 K=64 -> two 16x16x32 WMMAs
        v8f s = {};
#pragma unroll
        for (int c = 0; c < 2; ++c) {
            Frag a, b;
            const u16* ab = Qs + ln * DIMC + h * HD + c * 32 + hi * 8;
            a.q[0] = *(const u32x4*)(ab);
            a.q[1] = *(const u32x4*)(ab + 16);
            const u16* bb = Ks + ln * DIMC + h * HD + c * 32 + hi * 8;
            b.q[0] = *(const u32x4*)(bb);
            b.q[1] = *(const u32x4*)(bb + 16);
            s = wmma_bf16(a, b, s);
        }
        // softmax over keys: keys live across lanes within each 16-lane half
        float pr[8];
#pragma unroll
        for (int v = 0; v < 8; ++v) {
            float xv = s[v] * 0.125f;  // 1/sqrt(64)
            float mx = xv;
#pragma unroll
            for (int off = 8; off >= 1; off >>= 1)
                mx = fmaxf(mx, __shfl_xor(mx, off, 32));
            float p  = __expf(xv - mx);
            float sm = p;
#pragma unroll
            for (int off = 8; off >= 1; off >>= 1)
                sm += __shfl_xor(sm, off, 32);
            pr[v] = p / sm;
        }
        // transpose P (C-layout -> A-layout) via wave-private LDS bounce
        u16* pbase = Ps + wid * 256;
#pragma unroll
        for (int v = 0; v < 8; ++v) {
            const int m = v + hi * 8;
            pbase[m * 16 + ln] = f2bf(pr[v]);
        }
        __asm__ volatile("s_wait_dscnt 0x0" ::: "memory");
        Frag pa;
        {
            const u16* pp = pbase + ln * 16 + hi * 8;
            pa.q[0] = *(const u32x4*)(pp);   // K(keys) 0..7 / 8..15
            u32x4 z = {};
            pa.q[1] = z;                     // K 16..31 zero pad
        }
        // attn = P @ V : M=16 K=16(pad 32) N=64 -> four WMMAs
        v8f o[4];
#pragma unroll
        for (int j = 0; j < 4; ++j) o[j] = {};
#pragma unroll
        for (int j = 0; j < 4; ++j) {
            Frag vb;
            {  // lanes hold keys 8*hi..8*hi+7; upper 8 elements = zero pad
                const u16* vv = Vt + (h * HD + j * 16 + ln) * WIN;
                vb.q[0] = *(const u32x4*)(vv + hi * 8);
                u32x4 z = {};
                vb.q[1] = z;
            }
            o[j] = wmma_bf16(pa, vb, o[j]);
        }
#pragma unroll
        for (int j = 0; j < 4; ++j) {
#pragma unroll
            for (int v = 0; v < 8; ++v) {
                const int m = v + hi * 8;
                As[m * DIMC + h * HD + j * 16 + ln] = f2bf(o[j][v]);
            }
        }
    }
    __syncthreads();

    // ---- phase 5: out = attn @ wo^T -> global (fp32) ----
#pragma unroll
    for (int t = 0; t < 8; ++t) acc[t] = {};
    gemm_16x128(As, wo, nBase, ln, hi, acc);
#pragma unroll
    for (int t = 0; t < 8; ++t) {
#pragma unroll
        for (int v = 0; v < 8; ++v) {
            const int m = v + hi * 8;
            out[(size_t)(tokenBase + m) * DIMC + nBase + t * 16 + ln] = acc[t][v];
        }
    }
}

extern "C" void kernel_launch(void* const* d_in, const int* in_sizes, int n_in,
                              void* d_out, int out_size, void* d_ws, size_t ws_size,
                              hipStream_t stream) {
    const float* x    = (const float*)d_in[0];
    const float* rope = (const float*)d_in[1];
    const float* wq   = (const float*)d_in[2];
    const float* wk   = (const float*)d_in[3];
    const float* wv   = (const float*)d_in[4];
    const float* wo   = (const float*)d_in[5];
    float* out = (float*)d_out;

    const int total_tokens = in_sizes[0] / DIMC;   // B*S
    const int windows      = total_tokens / WIN;   // 1024

    // bf16 weight copies in workspace: 4 x [1024][1024] u16 = 8 MB
    u16* wq_bf = (u16*)d_ws;
    u16* wk_bf = wq_bf + (size_t)DIMC * DIMC;
    u16* wv_bf = wk_bf + (size_t)DIMC * DIMC;
    u16* wo_bf = wv_bf + (size_t)DIMC * DIMC;

    const int npairs = (DIMC * DIMC) / 2;          // 524288 packed pairs
    const int cblocks = (npairs + 255) / 256;
    cvt_w_kernel<<<cblocks, 256, 0, stream>>>(wq, (unsigned int*)wq_bf, npairs);
    cvt_w_kernel<<<cblocks, 256, 0, stream>>>(wk, (unsigned int*)wk_bf, npairs);
    cvt_w_kernel<<<cblocks, 256, 0, stream>>>(wv, (unsigned int*)wv_bf, npairs);
    cvt_w_kernel<<<cblocks, 256, 0, stream>>>(wo, (unsigned int*)wo_bf, npairs);

    // LDS: 4 * [16][1024] bf16 + P bounce + cos/sin tables
    const size_t shmem = (size_t)(4 * WIN * DIMC + 8 * 256) * sizeof(u16)
                       + (size_t)(2 * WIN * 32) * sizeof(float);  // = 139264 B
    hipFuncSetAttribute((const void*)lwa_fused_kernel,
                        hipFuncAttributeMaxDynamicSharedMemorySize, (int)shmem);
    lwa_fused_kernel<<<windows, 256, shmem, stream>>>(x, rope, wq_bf, wk_bf, wv_bf,
                                                      wo_bf, out);
}
